// DGCNN_56564719289094
// MI455X (gfx1250) — matrix operand order repaired and checked by reading, not compiled, not run on hardware
//
#include <hip/hip_runtime.h>
#include <hip/hip_bf16.h>
#include <cstdint>
#include <cstddef>

#define NPTS  1024
#define BATCH 8
#define KNN   20

typedef __attribute__((ext_vector_type(2)))  float    v2f;
typedef __attribute__((ext_vector_type(8)))  float    v8f;
typedef __attribute__((ext_vector_type(8)))  _Float16 v8h;
typedef __attribute__((ext_vector_type(16))) _Float16 v16h;

// ---------------------------------------------------------------------------
// x [B,N,3] -> xT [B,4,N] (channel 3 zero-padded so the f32 WMMA K=4 loop is exact)
__global__ __launch_bounds__(256) void xpose_kernel(const float* __restrict__ x,
                                                    float* __restrict__ xT) {
    int t = blockIdx.x * 256 + threadIdx.x;            // B*4*N threads
    int n = t % NPTS;
    int c = (t / NPTS) & 3;
    int b = t / (4 * NPTS);
    xT[t] = (c < 3) ? x[((long)b * NPTS + n) * 3 + c] : 0.0f;
}

// ---------------------------------------------------------------------------
// f32 [B,C,N] (strided view) -> f16 point-major [B,N,Cpad], zero pad K
__global__ __launch_bounds__(256) void tof16_kernel(const float* __restrict__ hin,
                                                    long bstride, int Cin, int Cpad,
                                                    _Float16* __restrict__ hT) {
    long t = blockIdx.x * 256L + threadIdx.x;          // B*N*Cpad threads
    int c = (int)(t % Cpad);
    long bn = t / Cpad;
    int n = (int)(bn % NPTS);
    int b = (int)(bn / NPTS);
    float v = (c < Cin) ? hin[(long)b * bstride + (long)c * NPTS + n] : 0.0f;
    hT[t] = (_Float16)v;
}

// ---------------------------------------------------------------------------
// xx[b,n] = sum_c h^2
__global__ __launch_bounds__(256) void xx_kernel(const float* __restrict__ hin,
                                                 int Cin, long bstride,
                                                 float* __restrict__ xx) {
    int t = blockIdx.x * 256 + threadIdx.x;            // B*N threads
    int n = t % NPTS;
    int b = t / NPTS;
    const float* hb = hin + (long)b * bstride;
    float s = 0.0f;
    for (int c = 0; c < Cin; ++c) { float v = hb[(long)c * NPTS + n]; s += v * v; }
    xx[t] = s;
}

// ---------------------------------------------------------------------------
// W [Cout, 2*Cin] f32 -> W1h = W[:, :Cin], W2h = W[:, Cin:] - W[:, :Cin], f16, K-padded
__global__ __launch_bounds__(256) void prepw_kernel(const float* __restrict__ W,
                                                    int Cin, int Cpad, int Cout,
                                                    _Float16* __restrict__ W1h,
                                                    _Float16* __restrict__ W2h) {
    int t = blockIdx.x * 256 + threadIdx.x;
    if (t >= Cout * Cpad) return;
    int c = t % Cpad;
    int o = t / Cpad;
    float w1 = 0.0f, w2 = 0.0f;
    if (c < Cin) {
        float a = W[(long)o * 2 * Cin + c];
        float b = W[(long)o * 2 * Cin + Cin + c];
        w1 = a;
        w2 = b - a;
    }
    W1h[t] = (_Float16)w1;
    W2h[t] = (_Float16)w2;
}

// ---------------------------------------------------------------------------
// Gram / neg-sqdist: S[b,i,j] = 2*<x_i,x_j> - xx_i - xx_j via V_WMMA_F32_16X16X4_F32.
// One wave per 16x64 strip (4 output tiles): A fragment loaded once per k-step,
// reused by 4 WMMAs -> 2.5 global loads per WMMA instead of 4.
__global__ __launch_bounds__(256) void gram_kernel(const float* __restrict__ h,
                                                   int C, long bstride,
                                                   const float* __restrict__ xx,
                                                   float* __restrict__ S) {
    int wave = (blockIdx.x * 256 + threadIdx.x) >> 5;
    int lane = threadIdx.x & 31;
    const int spb = 64 * 16;                            // strips per batch
    int b = wave / spb;
    int t = wave % spb;
    int i0 = (t >> 4) * 16;                             // row tile
    int j0 = (t & 15) * 64;                             // 4 adjacent col tiles
    const float* hb = h + (long)b * bstride;
    int lg = lane >> 4;                                 // lane group 0/1
    int lp = lane & 15;
    int ai = i0 + lp;                                   // A row (point i)
    v8f acc[4] = {};
    for (int k = 0; k < C; k += 4) {
        int ka = k + 2 * lg;
        const float* h0 = hb + (long)ka * NPTS;
        const float* h1 = h0 + NPTS;
        v2f a;
        a.x = h0[ai];
        a.y = h1[ai];
#pragma unroll
        for (int jt = 0; jt < 4; ++jt) {
            int bj = j0 + jt * 16 + lp;                 // B col (point j)
            v2f bb;
            bb.x = h0[bj];
            bb.y = h1[bj];
            acc[jt] = __builtin_amdgcn_wmma_f32_16x16x4_f32(false, a, false, bb,
                                                            (short)0, acc[jt],
                                                            false, false);
        }
    }
    const float* xxb = xx + b * NPTS;
#pragma unroll
    for (int jt = 0; jt < 4; ++jt) {
        float xj = xxb[j0 + jt * 16 + lp];
#pragma unroll
        for (int r = 0; r < 8; ++r) {
            int i = i0 + r + 8 * lg;                    // C/D layout: vgpr r, lane halves
            S[((long)b * NPTS + i) * NPTS + j0 + jt * 16 + lp] =
                2.0f * acc[jt][r] - xxb[i] - xj;
        }
    }
}

// ---------------------------------------------------------------------------
// Top-K=20 per row via repeated LDS argmax; tie-break = lowest index.
// The 4KB row is staged into LDS with GLOBAL_LOAD_ASYNC_TO_LDS_B128 (ASYNCcnt path,
// no VGPR bounce): 256 lanes x 16B = 1024 floats.
__global__ __launch_bounds__(256) void topk_kernel(const float* __restrict__ S,
                                                   int* __restrict__ idx) {
    __shared__ __align__(16) float vals[NPTS];
    __shared__ float bv[256];
    __shared__ int   bi[256];
    int row = blockIdx.x;                               // b*N + i
    int tid = threadIdx.x;
    const float* Sr = S + (long)row * NPTS;
#if defined(__gfx1250__)
    {
        unsigned int loff = (unsigned int)(uintptr_t)(&vals[tid * 4]); // LDS offset
        const float* gp = Sr + tid * 4;                 // 16B-aligned chunk
        asm volatile("global_load_async_to_lds_b128 %0, %1, off"
                     :: "v"(loff), "v"(gp) : "memory");
        asm volatile("s_wait_asynccnt 0x0" ::: "memory");
    }
#else
    for (int j = tid; j < NPTS; j += 256) vals[j] = Sr[j];
#endif
    __syncthreads();
    int* out = idx + (long)row * KNN;
    for (int t = 0; t < KNN; ++t) {
        float best = -3.402823e38f;
        int   bestj = NPTS;
        for (int j = tid; j < NPTS; j += 256) {
            float v = vals[j];
            if (v > best) { best = v; bestj = j; }      // strict > => lowest j wins per thread
        }
        bv[tid] = best; bi[tid] = bestj;
        __syncthreads();
        for (int s = 128; s > 0; s >>= 1) {
            if (tid < s) {
                float ov = bv[tid + s]; int oi = bi[tid + s];
                if (ov > bv[tid] || (ov == bv[tid] && oi < bi[tid])) {
                    bv[tid] = ov; bi[tid] = oi;
                }
            }
            __syncthreads();
        }
        if (tid == 0) { out[t] = bi[0]; vals[bi[0]] = -3.402823e38f; }
        __syncthreads();
    }
}

// ---------------------------------------------------------------------------
// U = W1h * h, V = W2h * h  (f16 in, f32 acc) via V_WMMA_F32_16X16X32_F16.
// One wave per 16(m) x 64(n) strip: A (weight) fragment reused by 4 WMMAs.
// A 16x32 f16 layout: lanes 0-15 row=lane: elems0-7=K0..7, elems8-15=K16..23;
//                     lanes16-31 row=lane-16: K8..15 / K24..31.
// B 32x16 f16 layout: lanes0-15 col=lane: elems0-7=K0..7, elems8-15=K16..23; lanes16-31: K8..15 / K24..31.
__global__ __launch_bounds__(256) void gemmUV_kernel(const _Float16* __restrict__ hT,
                                                     int Cpad,
                                                     const _Float16* __restrict__ W1h,
                                                     const _Float16* __restrict__ W2h,
                                                     int Cout,
                                                     float* __restrict__ U,
                                                     float* __restrict__ V) {
    int gw   = (blockIdx.x * 256 + threadIdx.x) >> 5;
    int lane = threadIdx.x & 31;
    int tm = Cout >> 4;
    int tpm = tm * 16;                                  // strips per matrix per batch
    int b = gw / (2 * tpm);
    int r = gw % (2 * tpm);
    int which = r / tpm;  r %= tpm;
    int m0 = (r / 16) * 16;
    int n0 = (r % 16) * 64;
    const _Float16* Wm = which ? W2h : W1h;
    float* D = (which ? V : U) + (long)b * Cout * NPTS;
    const _Float16* hb = hT + (long)b * NPTS * Cpad;
    int lg = lane >> 4;
    int lp = lane & 15;
    v8f acc[4] = {};
    for (int k = 0; k < Cpad; k += 32) {
        const _Float16* wrow = Wm + (long)(m0 + lp) * Cpad + k + lg * 8;
        v8h alo = *(const v8h*)(wrow);                  // K = k + lg*8 + [0..7]
        v8h ahi = *(const v8h*)(wrow + 16);             // K = k + 16 + lg*8 + [0..7]
        v16h a = __builtin_shufflevector(alo, ahi, 0, 1, 2, 3, 4, 5, 6, 7,
                                                   8, 9, 10, 11, 12, 13, 14, 15);
#pragma unroll
        for (int jt = 0; jt < 4; ++jt) {
            const _Float16* hrow =
                hb + (long)(n0 + jt * 16 + lp) * Cpad + k + lg * 8;
            v8h blo = *(const v8h*)(hrow);              // K = k + lg*8 + [0..7]
            v8h bhi = *(const v8h*)(hrow + 16);         // K = k + 16 + lg*8 + [0..7]
            v16h bm = __builtin_shufflevector(blo, bhi, 0, 1, 2, 3, 4, 5, 6, 7,
                                                        8, 9, 10, 11, 12, 13, 14, 15);
            acc[jt] = __builtin_amdgcn_wmma_f32_16x16x32_f16(false, a, false, bm,
                                                             (short)0, acc[jt],
                                                             false, false);
        }
    }
#pragma unroll
    for (int jt = 0; jt < 4; ++jt)
#pragma unroll
        for (int rr = 0; rr < 8; ++rr)
            D[(long)(m0 + rr + 8 * lg) * NPTS + n0 + jt * 16 + lp] = acc[jt][rr];
}

// ---------------------------------------------------------------------------
// y[b,o,n] = mean_k lrelu( (U[b,o,idx[b,n,k]] + V[b,o,n]) * scale_o + shift_o )
__global__ __launch_bounds__(256) void combine_kernel(const float* __restrict__ U,
                                                      const float* __restrict__ V,
                                                      const int* __restrict__ idx,
                                                      const float* __restrict__ g,
                                                      const float* __restrict__ be,
                                                      const float* __restrict__ mn,
                                                      const float* __restrict__ vr,
                                                      int Cout,
                                                      float* __restrict__ outB,
                                                      long outBStride, int chOff) {
    long t = blockIdx.x * 256L + threadIdx.x;           // B*Cout*N threads, n fastest
    int n = (int)(t % NPTS);
    int o = (int)((t / NPTS) % Cout);
    int b = (int)(t / ((long)NPTS * Cout));
    const int* id = idx + ((long)b * NPTS + n) * KNN;
    const float* Ub = U + ((long)b * Cout + o) * NPTS;
    float v  = V[((long)b * Cout + o) * NPTS + n];
    float sc = g[o] * rsqrtf(vr[o] + 1e-5f);
    float sh = be[o] - mn[o] * sc;
    float acc = 0.0f;
#pragma unroll
    for (int k = 0; k < KNN; ++k) {
        float z = (Ub[id[k]] + v) * sc + sh;
        acc += (z > 0.0f) ? z : 0.2f * z;
    }
    outB[(long)b * outBStride + (long)(chOff + o) * NPTS + n] = acc * (1.0f / KNN);
}

// ---------------------------------------------------------------------------
__global__ __launch_bounds__(256) void pool_kernel(const float* __restrict__ yf,
                                                   float* __restrict__ pooled) {
    int t = blockIdx.x * 256 + threadIdx.x;             // B*512 threads
    const float* p = yf + (long)t * NPTS;
    float s = 0.0f;
    for (int n = 0; n < NPTS; ++n) s += p[n];
    pooled[t] = s * (1.0f / NPTS);
}

__global__ __launch_bounds__(256) void linear_kernel(const float* __restrict__ pooled,
                                                     const float* __restrict__ We,
                                                     float* __restrict__ out) {
    int t = blockIdx.x * 256 + threadIdx.x;             // B*256 threads
    int j = t % 256;
    int b = t / 256;
    const float* p = pooled + (long)b * 512;
    const float* w = We + (long)j * 512;
    float s = 0.0f;
    for (int o = 0; o < 512; ++o) s += p[o] * w[o];
    out[t] = s;
}

// ---------------------------------------------------------------------------
extern "C" void kernel_launch(void* const* d_in, const int* in_sizes, int n_in,
                              void* d_out, int out_size, void* d_ws, size_t ws_size,
                              hipStream_t stream) {
    const float* x = (const float*)d_in[0];
    const float* Wl[4] = {(const float*)d_in[1],  (const float*)d_in[6],
                          (const float*)d_in[11], (const float*)d_in[16]};
    const float* gl[4] = {(const float*)d_in[2],  (const float*)d_in[7],
                          (const float*)d_in[12], (const float*)d_in[17]};
    const float* bl[4] = {(const float*)d_in[3],  (const float*)d_in[8],
                          (const float*)d_in[13], (const float*)d_in[18]};
    const float* ml[4] = {(const float*)d_in[4],  (const float*)d_in[9],
                          (const float*)d_in[14], (const float*)d_in[19]};
    const float* vl[4] = {(const float*)d_in[5],  (const float*)d_in[10],
                          (const float*)d_in[15], (const float*)d_in[20]};
    const float* We = (const float*)d_in[21];
    float* out = (float*)d_out;

    char* ws = (char*)d_ws;
    size_t off = 0;
    auto carve = [&](size_t bytes) {
        char* p = ws + off;
        off = (off + bytes + 255) & ~(size_t)255;
        return p;
    };
    float*    xT     = (float*)   carve((size_t)BATCH * 4 * NPTS * 4);
    _Float16* hT16   = (_Float16*)carve((size_t)BATCH * NPTS * 448 * 2);
    float*    xxb    = (float*)   carve((size_t)BATCH * NPTS * 4);
    _Float16* W1h    = (_Float16*)carve((size_t)512 * 448 * 2);
    _Float16* W2h    = (_Float16*)carve((size_t)512 * 448 * 2);
    float*    Sreg   = (float*)   carve((size_t)BATCH * NPTS * NPTS * 4); // aliased by U,V
    int*      idxbuf = (int*)     carve((size_t)BATCH * NPTS * KNN * 4);
    float*    hcat   = (float*)   carve((size_t)BATCH * 448 * NPTS * 4);
    float*    yf     = (float*)   carve((size_t)BATCH * 512 * NPTS * 4);
    float*    pooled = (float*)   carve((size_t)BATCH * 512 * 4);
    float* U = Sreg;                                    // Gram dead after topk; exact fit
    float* V = Sreg + (size_t)BATCH * 512 * NPTS;

    xpose_kernel<<<(BATCH * 4 * NPTS) / 256, 256, 0, stream>>>(x, xT);

    struct Cfg { const float* hin; long bstride; int Cin, Cgram, Cpad, Cout;
                 float* outB; long outBStride; int chOff; };
    const Cfg cfg[4] = {
        { xT,                4L * NPTS,   3,   4,   32,  64,  hcat, 448L * NPTS, 0   },
        { hcat,              448L * NPTS, 64,  64,  64,  128, hcat, 448L * NPTS, 64  },
        { hcat + 64 * NPTS,  448L * NPTS, 128, 128, 128, 256, hcat, 448L * NPTS, 192 },
        { hcat,              448L * NPTS, 448, 448, 448, 512, yf,   512L * NPTS, 0   },
    };

    for (int l = 0; l < 4; ++l) {
        const Cfg& c = cfg[l];
        tof16_kernel<<<(BATCH * NPTS * c.Cpad) / 256, 256, 0, stream>>>(
            c.hin, c.bstride, c.Cin, c.Cpad, hT16);
        xx_kernel<<<(BATCH * NPTS) / 256, 256, 0, stream>>>(c.hin, c.Cin, c.bstride, xxb);
        prepw_kernel<<<(c.Cout * c.Cpad + 255) / 256, 256, 0, stream>>>(
            Wl[l], c.Cin, c.Cpad, c.Cout, W1h, W2h);
        gram_kernel<<<(BATCH * 64 * 16) / 8, 256, 0, stream>>>(
            c.hin, c.Cgram, c.bstride, xxb, Sreg);
        topk_kernel<<<BATCH * NPTS, 256, 0, stream>>>(Sreg, idxbuf);
        gemmUV_kernel<<<(BATCH * 2 * (c.Cout / 16) * 16) / 8, 256, 0, stream>>>(
            hT16, c.Cpad, W1h, W2h, c.Cout, U, V);
        combine_kernel<<<(int)(((long)BATCH * c.Cout * NPTS) / 256), 256, 0, stream>>>(
            U, V, idxbuf, gl[l], bl[l], ml[l], vl[l], c.Cout, c.outB, c.outBStride, c.chOff);
    }

    pool_kernel<<<(BATCH * 512) / 256, 256, 0, stream>>>(yf, pooled);
    linear_kernel<<<(BATCH * 256) / 256, 256, 0, stream>>>(pooled, We, out);
}
